// Head_249108103428
// MI455X (gfx1250) — matrix operand order repaired and checked by reading, not compiled, not run on hardware
//
#include <hip/hip_runtime.h>
#include <hip/hip_bf16.h>
#include <stdint.h>

#define EMB 1024
#define HD  64
#define BB  8
#define TT  2048

typedef __attribute__((ext_vector_type(16))) __bf16 v16bf;
typedef __attribute__((ext_vector_type(8)))  __bf16 v8bf;
typedef __attribute__((ext_vector_type(8)))  float  v8f;
typedef __attribute__((ext_vector_type(4)))  float  v4f;
typedef __attribute__((ext_vector_type(4)))  unsigned int u32x4;
typedef __attribute__((ext_vector_type(8)))  int    i32x8;
typedef __attribute__((ext_vector_type(4)))  int    i32x4;

#if __has_builtin(__builtin_amdgcn_tensor_load_to_lds)
#define USE_TDM 1
#else
#define USE_TDM 0
#endif

static __device__ __forceinline__ v8f wmma_bf16(v16bf a, v16bf b, v8f c) {
  return __builtin_amdgcn_wmma_f32_16x16x32_bf16(false, a, false, b, (short)0, c,
                                                 false, false);
}

// ---- Tensor Data Mover: 2D tile (bf16) global -> LDS, per CDNA5 ISA D# layout ----
// group0: [0]=count=1, [1]=lds_addr, [2]=ga[31:0], [3]=ga[56:32]|type2<<30
// group1: d0 = data_size(2B)<<16 ; dim0 @79:48 ; dim1 @111:80 ; tile0 @127:112 ;
//         tile1 @143:128 ; dim0_stride @207:160 ; rest 0.
static __device__ __forceinline__ void tdm_load_2d(
    __bf16* lds_dst, const __bf16* gsrc,
    uint32_t tensor_d0, uint32_t tensor_d1,
    uint32_t tile_d0, uint32_t tile_d1, uint32_t d0_stride) {
#if USE_TDM
  const uint64_t ga = (uint64_t)(uintptr_t)gsrc;
  u32x4 g0;
  g0[0] = 1u;                                                // count=1 (valid user D#)
  g0[1] = (uint32_t)(uintptr_t)lds_dst;                      // LDS byte address
  g0[2] = (uint32_t)ga;                                      // global_addr[31:0]
  g0[3] = (uint32_t)((ga >> 32) & 0x01FFFFFFu) | (2u << 30); // ga[56:32] | type=2
  i32x8 g1;
  g1[0] = 0x00010000;                                        // data_size=1 -> 2 bytes
  g1[1] = (int)((tensor_d0 & 0xFFFFu) << 16);                // dim0[15:0] -> bits 63:48
  g1[2] = (int)((tensor_d0 >> 16) | ((tensor_d1 & 0xFFFFu) << 16));
  g1[3] = (int)((tensor_d1 >> 16) | (tile_d0 << 16));        // dim1[31:16] | tile_dim0
  g1[4] = (int)tile_d1;                                      // tile_dim1 (tile_dim2=0)
  g1[5] = (int)d0_stride;                                    // dim0_stride[31:0]
  g1[6] = 0;                                                 // stride[47:32]=0, dim1_stride=0
  g1[7] = 0;
  const i32x4 z4 = {0, 0, 0, 0};
  const i32x8 z8 = {0, 0, 0, 0, 0, 0, 0, 0};
  __builtin_amdgcn_tensor_load_to_lds(g0, g1, z4, z4, z8, 0);
#else
  const int l = threadIdx.x & 31;
  for (int r = l; r < (int)tile_d1; r += 32)
    for (int c = 0; c < (int)tile_d0; c += 8)
      *(v8bf*)(lds_dst + (size_t)r * tile_d0 + c) =
          *(const v8bf*)(gsrc + (size_t)r * d0_stride + c);
#endif
}

static __device__ __forceinline__ void tdm_wait() {
#if USE_TDM
  __builtin_amdgcn_s_wait_tensorcnt((short)0);
#endif
}

// A-matrix 16x32 bf16 fragment from an f32 row-major tile at k-offset kk.
static __device__ __forceinline__ v16bf load_a_f32(const float* base, int ld, int kk) {
  const int l = threadIdx.x & 31;
  const int row = l & 15, h = l >> 4;
  const float* p = base + (size_t)row * ld + kk + 16 * h;
  const v4f x0 = *(const v4f*)(p);
  const v4f x1 = *(const v4f*)(p + 4);
  const v4f x2 = *(const v4f*)(p + 8);
  const v4f x3 = *(const v4f*)(p + 12);
  v16bf a;
#pragma unroll
  for (int j = 0; j < 4; ++j) {
    a[j]      = (__bf16)x0[j];
    a[j + 4]  = (__bf16)x1[j];
    a[j + 8]  = (__bf16)x2[j];
    a[j + 12] = (__bf16)x3[j];
  }
  return a;
}

// Same A layout from a bf16 row-major tile (global or LDS); two 16B vector loads.
static __device__ __forceinline__ v16bf load_a_bf16(const __bf16* base, int ld, int kk) {
  const int l = threadIdx.x & 31;
  const int row = l & 15, h = l >> 4;
  const __bf16* p = base + (size_t)row * ld + kk + 16 * h;
  const v8bf lo = *(const v8bf*)(p);
  const v8bf hi = *(const v8bf*)(p + 8);
  return __builtin_shufflevector(lo, hi, 0, 1, 2, 3, 4, 5, 6, 7,
                                 8, 9, 10, 11, 12, 13, 14, 15);
}

// B fragment from pre-packed W: one contiguous 32B load per lane.
static __device__ __forceinline__ v16bf load_b_packed(const __bf16* wp, int kchunk, int t) {
  const int l = threadIdx.x & 31;
  return *(const v16bf*)(wp + ((((size_t)kchunk * 4 + t) * 32 + l) << 4));
}

// B fragment for Q*K^T from a row-major K tile: B[k][n] = Kt[row0+n][k].
static __device__ __forceinline__ v16bf load_b_rowT(const __bf16* kt, int row0, int kk) {
  const int l = threadIdx.x & 31;
  const int n = l & 15;
  const int kb = kk + ((l >> 4) << 4);
  return *(const v16bf*)(kt + (size_t)(row0 + n) * HD + kb);
}

// B fragment for P*V from the LDS V^T tile [64 head rows][32 keys].
static __device__ __forceinline__ v16bf load_b_vtile(const __bf16* vt, int n0) {
  const int l = threadIdx.x & 31;
  const int n = l & 15;
  const int kb = (l >> 4) << 4;
  return *(const v16bf*)(vt + (size_t)(n0 + n) * 32 + kb);
}

// ------------- Phase 0: pack W (f32 row-major) into bf16 B-fragment layout -------------
__global__ void __launch_bounds__(128) pack_w_kernel(
    const float* __restrict__ Wk, const float* __restrict__ Wq, const float* __restrict__ Wv,
    __bf16* __restrict__ pk, __bf16* __restrict__ pq, __bf16* __restrict__ pv) {
  const int tid = blockIdx.x * blockDim.x + threadIdx.x;  // 3 * 4096 entries
  const int m = tid >> 12;
  const int r = tid & 4095;            // ((kchunk*4 + t)*32 + lane)
  const int kchunk = r >> 7;
  const int t = (r >> 5) & 3;
  const int lane = r & 31;
  const int n = lane & 15;
  const int kb = kchunk * 32 + ((lane >> 4) << 4);
  const float* w = (m == 0) ? Wk : (m == 1) ? Wq : Wv;
  __bf16* dst = (m == 0) ? pk : (m == 1) ? pq : pv;
  v16bf b;
#pragma unroll
  for (int j = 0; j < 16; ++j)
    b[j] = (__bf16)w[(size_t)(kb + j) * HD + t * 16 + n];
  *(v16bf*)(dst + ((size_t)r << 4)) = b;
}

// ---------------- Phase 1: fused QKV projection (x read exactly once) ----------------
__global__ void __launch_bounds__(128) qkv_proj_kernel(
    const float* __restrict__ x,
    const __bf16* __restrict__ pk, const __bf16* __restrict__ pq, const __bf16* __restrict__ pv,
    __bf16* __restrict__ qb, __bf16* __restrict__ kb, __bf16* __restrict__ vt) {
  const int wave = blockIdx.x * 4 + (threadIdx.x >> 5);
  const int row0 = wave << 4;
  const int l = threadIdx.x & 31;
  const int n = l & 15, h = l >> 4;
  const float* xr = x + (size_t)row0 * EMB;

  v8f aq[4] = {}, ak[4] = {}, av[4] = {};
  for (int kk = 0; kk < EMB; kk += 32) {
    const int kc = kk >> 5;
    v16bf a = load_a_f32(xr, EMB, kk);
#pragma unroll
    for (int t = 0; t < 4; ++t) {
      aq[t] = wmma_bf16(a, load_b_packed(pq, kc, t), aq[t]);
      ak[t] = wmma_bf16(a, load_b_packed(pk, kc, t), ak[t]);
      av[t] = wmma_bf16(a, load_b_packed(pv, kc, t), av[t]);
    }
  }
  const float qscale = 0.03125f;  // 1024^-0.5 folded into Q
#pragma unroll
  for (int t = 0; t < 4; ++t) {
#pragma unroll
    for (int r = 0; r < 8; ++r) {
      const int row = row0 + r + 8 * h;
      const int col = t * 16 + n;
      const size_t idx = (size_t)row * HD + col;
      qb[idx] = (__bf16)(aq[t][r] * qscale);
      kb[idx] = (__bf16)ak[t][r];
      vt[((size_t)(row >> 11) * HD + col) * TT + (row & (TT - 1))] = (__bf16)av[t][r];
    }
  }
}

// -------- Phase 2: flash attention; TDM double-buffers K/V^T tiles into LDS --------
__global__ void __launch_bounds__(128) flash_attn_kernel(
    const __bf16* __restrict__ qb, const __bf16* __restrict__ kbg,
    const __bf16* __restrict__ vtb, const unsigned char* __restrict__ mask,
    float* __restrict__ out) {
  __shared__ __attribute__((aligned(128))) __bf16 kls[2][32 * HD];  // [32 keys][64]
  __shared__ __attribute__((aligned(128))) __bf16 vls[2][HD * 32];  // [64 head][32 keys]
  __shared__ __attribute__((aligned(64)))  __bf16 plds[4][16 * 32];

  const int wid = threadIdx.x >> 5;
  const int l = threadIdx.x & 31;
  const int n = l & 15, h = l >> 4;
  const int b    = blockIdx.x >> 5;    // 32 query-blocks of 64 rows per batch
  const int qblk = blockIdx.x & 31;
  const int q0 = qblk * 64 + wid * 16;

  const __bf16* Q  = qb  + ((size_t)b * TT + q0) * HD;
  const __bf16* K  = kbg + (size_t)b * TT * HD;
  const __bf16* Vt = vtb + (size_t)b * HD * TT;
  const unsigned char* mk = mask + (size_t)b * TT;

  const v16bf qa0 = load_a_bf16(Q, HD, 0);
  const v16bf qa1 = load_a_bf16(Q, HD, 32);

  v8f o[4] = {};
  float m[8], ls[8];
#pragma unroll
  for (int r = 0; r < 8; ++r) { m[r] = -1e30f; ls[r] = 0.f; }

  __bf16* pw = plds[wid];
  const int nkb_w   = (q0 + 47) >> 5;               // this wave's causal key blocks
  const int nkb_blk = (qblk * 64 + 48 + 47) >> 5;   // block max (uniform)

  if (wid == 0) {  // prologue DMA: key block 0 into buffer 0
    tdm_load_2d(kls[0], K,  HD, TT, HD, 32, HD);
    tdm_load_2d(vls[0], Vt, TT, HD, 32, HD, TT);
  }

  for (int kblk = 0; kblk < nkb_blk; ++kblk) {
    const int cur = kblk & 1;
    if (wid == 0) tdm_wait();          // wave 0 owns TENSORcnt
    __syncthreads();                   // tiles[cur] visible to all waves
    if (wid == 0 && kblk + 1 < nkb_blk) {
      const int knext = (kblk + 1) << 5;
      tdm_load_2d(kls[1 - cur], K + (size_t)knext * HD, HD, TT - knext, HD, 32, HD);
      tdm_load_2d(vls[1 - cur], Vt + knext, TT - knext, HD, 32, HD, TT);
    }

    if (kblk < nkb_w) {                // wave-uniform: EXEC stays all-ones for WMMA
      const int key0 = kblk << 5;
      const __bf16* kt = kls[cur];
      const __bf16* vt = vls[cur];

      // S = Q * K^T over 32 keys -> two 16x16 f32 tiles (from LDS K tile)
      v8f s[2];
#pragma unroll
      for (int t = 0; t < 2; ++t) {
        v8f acc = {};
        acc = wmma_bf16(qa0, load_b_rowT(kt, 16 * t, 0),  acc);
        acc = wmma_bf16(qa1, load_b_rowT(kt, 16 * t, 32), acc);
        s[t] = acc;
      }

      // causal + attn mask; running per-row max (row = r + 8h, col = key0 + 16t + n)
      float mn[8];
#pragma unroll
      for (int r = 0; r < 8; ++r) mn[r] = m[r];
#pragma unroll
      for (int t = 0; t < 2; ++t) {
        const int kc = key0 + 16 * t + n;
        const bool mok = mk[kc] != 0;
#pragma unroll
        for (int r = 0; r < 8; ++r) {
          const int qr = q0 + r + 8 * h;
          float sv = s[t][r];
          sv = (kc <= qr && mok) ? sv : -1e30f;
          s[t][r] = sv;
          mn[r] = fmaxf(mn[r], sv);
        }
      }
#pragma unroll
      for (int r = 0; r < 8; ++r) {
#pragma unroll
        for (int off = 1; off < 16; off <<= 1)
          mn[r] = fmaxf(mn[r], __shfl_xor(mn[r], off, 32));
      }

      float ps[8];
#pragma unroll
      for (int r = 0; r < 8; ++r) {
        const float sc = __expf(m[r] - mn[r]);
        m[r] = mn[r];
        ls[r] *= sc;
        ps[r] = 0.f;
#pragma unroll
        for (int t = 0; t < 4; ++t) o[t][r] *= sc;
      }

      // P = exp(S - m); stage bf16 P into per-wave LDS for the A-fragment reload
#pragma unroll
      for (int t = 0; t < 2; ++t) {
#pragma unroll
        for (int r = 0; r < 8; ++r) {
          const float sv = s[t][r];
          const float p = (sv < -1e29f) ? 0.f : __expf(sv - m[r]);
          ps[r] += p;
          pw[(r + 8 * h) * 32 + 16 * t + n] = (__bf16)p;
        }
      }
#pragma unroll
      for (int r = 0; r < 8; ++r) {
        float t_ = ps[r];
#pragma unroll
        for (int off = 1; off < 16; off <<= 1) t_ += __shfl_xor(t_, off, 32);
        ls[r] += t_;
      }

      // O += P * V  (A = 16x32 P from LDS, B = V^T LDS tile, 4 head tiles)
      const v16bf pa = load_a_bf16(pw, 32, 0);
#pragma unroll
      for (int t = 0; t < 4; ++t)
        o[t] = wmma_bf16(pa, load_b_vtile(vt, 16 * t), o[t]);
    }
  }

  // normalize and store f32 output
#pragma unroll
  for (int r = 0; r < 8; ++r) {
    const float inv = (ls[r] > 0.f) ? 1.f / ls[r] : 0.f;
    const int qr = q0 + r + 8 * h;
    float* op = out + ((size_t)b * TT + qr) * HD;
#pragma unroll
    for (int t = 0; t < 4; ++t) op[t * 16 + n] = o[t][r] * inv;
  }
}

extern "C" void kernel_launch(void* const* d_in, const int* in_sizes, int n_in,
                              void* d_out, int out_size, void* d_ws, size_t ws_size,
                              hipStream_t stream) {
  (void)in_sizes; (void)n_in; (void)out_size; (void)ws_size;
  const float*         x    = (const float*)d_in[0];
  const unsigned char* mask = (const unsigned char*)d_in[1];  // jnp bool -> 1 byte
  const float*         Wk   = (const float*)d_in[2];
  const float*         Wq   = (const float*)d_in[3];
  const float*         Wv   = (const float*)d_in[4];
  float*               out  = (float*)d_out;

  const size_t nqkv  = (size_t)BB * TT * HD;    // 1M elems per matrix
  const size_t npack = (size_t)EMB * HD;        // 64K elems per packed W
  __bf16* qb  = (__bf16*)d_ws;                  // 2 MB   Q (row-major, pre-scaled)
  __bf16* kbf = qb + nqkv;                      // 2 MB   K (row-major)
  __bf16* vt  = kbf + nqkv;                     // 2 MB   V transposed [B][HD][T]
  __bf16* pwk = vt + nqkv;                      // 128 KB packed Wk
  __bf16* pwq = pwk + npack;                    // 128 KB packed Wq
  __bf16* pwv = pwq + npack;                    // 128 KB packed Wv

  pack_w_kernel<<<(3 * 4096) / 128, 128, 0, stream>>>(Wk, Wq, Wv, pwk, pwq, pwv);
  qkv_proj_kernel<<<(BB * TT / 16) / 4, 128, 0, stream>>>(x, pwk, pwq, pwv, qb, kbf, vt);
  flash_attn_kernel<<<BB * (TT / 64), 128, 0, stream>>>(qb, kbf, vt, mask, out);
}
